// AttentionHead_10041633538725
// MI455X (gfx1250) — matrix operand order repaired and checked by reading, not compiled, not run on hardware
//
#include <hip/hip_runtime.h>
#include <hip/hip_bf16.h>

typedef _Float16 v16h __attribute__((ext_vector_type(16)));
typedef _Float16 half8 __attribute__((ext_vector_type(8)));
typedef float    v8f  __attribute__((ext_vector_type(8)));

#define DD 1024
#define NN 128
#define SS 4096
#define BB 4

union Frag16 { v16h v; half8 h[2]; };

__device__ __forceinline__ v16h ld_frag2(const _Float16* p0, const _Float16* p1) {
    Frag16 f;
    f.h[0] = *(const half8*)p0;
    f.h[1] = *(const half8*)p1;
    return f.v;
}

__device__ __forceinline__ v16h cvt_a_frag(const float* base) {
    // halves 0..7 = base[0..7], halves 8..15 = base[16..23] (A pattern, hi*8 pre-applied)
    v16h a;
    const float4* p0 = (const float4*)base;
    const float4* p1 = (const float4*)(base + 16);
    float4 x0 = p0[0], x1 = p0[1], x2 = p1[0], x3 = p1[1];
    a[0]=(_Float16)x0.x; a[1]=(_Float16)x0.y; a[2]=(_Float16)x0.z; a[3]=(_Float16)x0.w;
    a[4]=(_Float16)x1.x; a[5]=(_Float16)x1.y; a[6]=(_Float16)x1.z; a[7]=(_Float16)x1.w;
    a[8]=(_Float16)x2.x; a[9]=(_Float16)x2.y; a[10]=(_Float16)x2.z; a[11]=(_Float16)x2.w;
    a[12]=(_Float16)x3.x; a[13]=(_Float16)x3.y; a[14]=(_Float16)x3.z; a[15]=(_Float16)x3.w;
    return a;
}

__device__ __forceinline__ v16h cvt_b_frag(const float* base) {
    // halves 0..7 = base[0..7], halves 8..15 = base[16..23] (B pattern, hi*8 pre-applied)
    v16h b;
    const float4* p = (const float4*)base;
    float4 y0 = p[0], y1 = p[1], y2 = p[4], y3 = p[5];
    b[0]=(_Float16)y0.x; b[1]=(_Float16)y0.y; b[2]=(_Float16)y0.z; b[3]=(_Float16)y0.w;
    b[4]=(_Float16)y1.x; b[5]=(_Float16)y1.y; b[6]=(_Float16)y1.z; b[7]=(_Float16)y1.w;
    b[8]=(_Float16)y2.x; b[9]=(_Float16)y2.y; b[10]=(_Float16)y2.z; b[11]=(_Float16)y2.w;
    b[12]=(_Float16)y3.x; b[13]=(_Float16)y3.y; b[14]=(_Float16)y3.z; b[15]=(_Float16)y3.w;
    return b;
}

// ---------------------------------------------------------------------------
// Kernel 1: fused q/k/v projection.  One wave computes one 16x16 tile.
// Two independent accumulator chains to fill WMMA hazard slots.
// ---------------------------------------------------------------------------
__global__ __launch_bounds__(256) void proj_kernel(
    const float* __restrict__ x,
    const float* __restrict__ Wq, const float* __restrict__ Wk,
    const float* __restrict__ Wv,
    _Float16* __restrict__ q, _Float16* __restrict__ k,
    _Float16* __restrict__ vt)
{
    const int lane = threadIdx.x & 31;
    const int wave = threadIdx.x >> 5;
    const int task = blockIdx.x * 8 + wave;          // 0 .. 24575
    const int mtile = task / 24;
    const int sel   = task % 24;
    const int row0  = mtile * 16;                    // global row over B*S
    const int r  = lane & 15;
    const int hi = lane >> 4;

    const float* W;
    int nt;
    if (sel < 8)       { W = Wq; nt = sel; }
    else if (sel < 16) { W = Wk; nt = sel - 8; }
    else               { W = Wv; nt = sel - 16; }

    const float* arow = x + (size_t)(row0 + r) * DD + hi * 8;
    const float* brow = W + (size_t)(nt * 16 + r) * DD + hi * 8;

    v8f acc0 = {}, acc1 = {};
#pragma unroll 2
    for (int k0 = 0; k0 < DD; k0 += 64) {
        v16h a0 = cvt_a_frag(arow + k0);
        v16h b0 = cvt_b_frag(brow + k0);
        v16h a1 = cvt_a_frag(arow + k0 + 32);
        v16h b1 = cvt_b_frag(brow + k0 + 32);
        acc0 = __builtin_amdgcn_wmma_f32_16x16x32_f16(
            false, a0, false, b0, (short)0, acc0, false, false);
        acc1 = __builtin_amdgcn_wmma_f32_16x16x32_f16(
            false, a1, false, b1, (short)0, acc1, false, false);
    }
    v8f acc = acc0 + acc1;

    if (sel < 16) {
        _Float16* out = (sel < 8) ? q : k;
        _Float16* o = out + (size_t)(row0 + 8 * hi) * NN + nt * 16 + r;
#pragma unroll
        for (int vr = 0; vr < 8; ++vr)
            o[(size_t)vr * NN] = (_Float16)acc[vr];
    } else {
        const int bidx = row0 >> 12;
        const int s    = row0 & (SS - 1);
        half8 h;
#pragma unroll
        for (int vr = 0; vr < 8; ++vr) h[vr] = (_Float16)acc[vr];
        _Float16* o = vt + ((size_t)bidx * NN + nt * 16 + r) * SS + s + 8 * hi;
        *(half8*)o = h;
    }
}

// ---------------------------------------------------------------------------
// Online-softmax row update for one 16x32 score tile held as two C-layout v8f.
// m/ss/corr are per-row stats replicated across 16-lane halves.
// ---------------------------------------------------------------------------
__device__ __forceinline__ void softmax_update(
    v8f& s0, v8f& s1, float* m, float* ss, float* corr, float scale)
{
#pragma unroll
    for (int i = 0; i < 8; ++i) {
        float a0 = s0[i] * scale, a1 = s1[i] * scale;
        float mx = fmaxf(a0, a1);
        mx = fmaxf(mx, __shfl_xor(mx, 1, 16));
        mx = fmaxf(mx, __shfl_xor(mx, 2, 16));
        mx = fmaxf(mx, __shfl_xor(mx, 4, 16));
        mx = fmaxf(mx, __shfl_xor(mx, 8, 16));
        float mn = fmaxf(m[i], mx);
        corr[i] = __expf(m[i] - mn);
        m[i] = mn;
        float p0 = __expf(a0 - mn), p1 = __expf(a1 - mn);
        s0[i] = p0; s1[i] = p1;
        float rs = p0 + p1;
        rs += __shfl_xor(rs, 1, 16);
        rs += __shfl_xor(rs, 2, 16);
        rs += __shfl_xor(rs, 4, 16);
        rs += __shfl_xor(rs, 8, 16);
        ss[i] = ss[i] * corr[i] + rs;
    }
}

// ---------------------------------------------------------------------------
// Kernel 2: flash-attention.  One wave per 32-query tile (two 16-row M-tiles
// sharing every K/V fragment load -> 2x WMMA per vmem op, 4 independent
// score-accumulator chains).  Online softmax over key-blocks of 32.
// ---------------------------------------------------------------------------
__global__ __launch_bounds__(32) void attn_kernel(
    const _Float16* __restrict__ q, const _Float16* __restrict__ k,
    const _Float16* __restrict__ vt, float* __restrict__ out)
{
    __shared__ _Float16 lp[2 * 16 * 32];

    const int lane = threadIdx.x & 31;
    const int r  = lane & 15;
    const int hi = lane >> 4;
    const int qt = blockIdx.x;                       // 0..511
    const int b  = qt >> 7;                          // 128 qtiles of 32 per batch
    const int s0 = (qt & 127) * 32;

    const _Float16* qb = q  + (size_t)b * SS * NN;
    const _Float16* kb = k  + (size_t)b * SS * NN;
    const _Float16* vb = vt + (size_t)b * NN * SS;

    // resident Q fragments: 32 x 128 as two M-tiles x 4 K-chunks
    v16h qa[8];
#pragma unroll
    for (int t = 0; t < 2; ++t)
#pragma unroll
        for (int d = 0; d < 4; ++d) {
            const _Float16* p =
                qb + (size_t)(s0 + t * 16 + r) * NN + d * 32 + hi * 8;
            qa[t * 4 + d] = ld_frag2(p, p + 16);
        }

    v8f O[16];
    float m[16], ss[16];
#pragma unroll
    for (int n = 0; n < 16; ++n) O[n] = v8f{};
#pragma unroll
    for (int i = 0; i < 16; ++i) { m[i] = -1e30f; ss[i] = 0.0f; }

    const float scale = 0.03125f;                    // 1/sqrt(1024)

    for (int kb0 = 0; kb0 < SS; kb0 += 32) {
        // scores: two 16x32 tiles (4 independent WMMA chains)
        v8f sc00 = {}, sc01 = {}, sc10 = {}, sc11 = {};
#pragma unroll
        for (int d = 0; d < 4; ++d) {
            const _Float16* p0 = kb + (size_t)(kb0 + r) * NN + d * 32 + hi * 8;
            const _Float16* p1 = kb + (size_t)(kb0 + 16 + r) * NN + d * 32 + hi * 8;
            v16h b0 = ld_frag2(p0, p0 + 16);
            v16h b1 = ld_frag2(p1, p1 + 16);
            sc00 = __builtin_amdgcn_wmma_f32_16x16x32_f16(
                false, qa[d], false, b0, (short)0, sc00, false, false);
            sc01 = __builtin_amdgcn_wmma_f32_16x16x32_f16(
                false, qa[d], false, b1, (short)0, sc01, false, false);
            sc10 = __builtin_amdgcn_wmma_f32_16x16x32_f16(
                false, qa[4 + d], false, b0, (short)0, sc10, false, false);
            sc11 = __builtin_amdgcn_wmma_f32_16x16x32_f16(
                false, qa[4 + d], false, b1, (short)0, sc11, false, false);
        }

        float corr[16];
        softmax_update(sc00, sc01, m,     ss,     corr,     scale);
        softmax_update(sc10, sc11, m + 8, ss + 8, corr + 8, scale);

        // P tiles: C-layout -> LDS (row-major 16x32 f16, one per M-tile)
#pragma unroll
        for (int i = 0; i < 8; ++i) {
            const int row = i + 8 * hi;
            lp[row * 32 + r]              = (_Float16)sc00[i];
            lp[row * 32 + 16 + r]         = (_Float16)sc01[i];
            lp[512 + row * 32 + r]        = (_Float16)sc10[i];
            lp[512 + row * 32 + 16 + r]   = (_Float16)sc11[i];
        }
        __syncthreads();
        const _Float16* pp0 = lp + r * 32 + hi * 8;
        const _Float16* pp1 = lp + 512 + r * 32 + hi * 8;
        v16h pa0 = ld_frag2(pp0, pp0 + 16);
        v16h pa1 = ld_frag2(pp1, pp1 + 16);
        __syncthreads();

        // O = O*corr + P x V   (one V fragment feeds both M-tiles)
#pragma unroll
        for (int n = 0; n < 8; ++n) {
            const _Float16* vp = vb + (size_t)(n * 16 + r) * SS + kb0 + hi * 8;
            v16h bf = ld_frag2(vp, vp + 16);
#pragma unroll
            for (int i = 0; i < 8; ++i) O[n][i]     *= corr[i];
#pragma unroll
            for (int i = 0; i < 8; ++i) O[8 + n][i] *= corr[8 + i];
            O[n] = __builtin_amdgcn_wmma_f32_16x16x32_f16(
                false, pa0, false, bf, (short)0, O[n], false, false);
            O[8 + n] = __builtin_amdgcn_wmma_f32_16x16x32_f16(
                false, pa1, false, bf, (short)0, O[8 + n], false, false);
        }
    }

    // epilogue: divide by row sums, store f32
    float inv[16];
#pragma unroll
    for (int i = 0; i < 16; ++i) inv[i] = 1.0f / ss[i];
#pragma unroll
    for (int t = 0; t < 2; ++t)
#pragma unroll
        for (int n = 0; n < 8; ++n) {
            float* o = out + ((size_t)b * SS + s0 + t * 16 + 8 * hi) * NN
                           + n * 16 + r;
#pragma unroll
            for (int i = 0; i < 8; ++i)
                o[(size_t)i * NN] = O[t * 8 + n][i] * inv[t * 8 + i];
        }
}

// ---------------------------------------------------------------------------
extern "C" void kernel_launch(void* const* d_in, const int* in_sizes, int n_in,
                              void* d_out, int out_size, void* d_ws, size_t ws_size,
                              hipStream_t stream)
{
    const float* x  = (const float*)d_in[0];
    const float* Wq = (const float*)d_in[1];
    const float* Wk = (const float*)d_in[2];
    const float* Wv = (const float*)d_in[3];
    float* out = (float*)d_out;

    // workspace: q (4MB) | k (4MB) | v^T (4MB), all f16
    const size_t qkv_elems = (size_t)BB * SS * NN;   // 2,097,152
    _Float16* wq16 = (_Float16*)d_ws;
    _Float16* wk16 = wq16 + qkv_elems;
    _Float16* wvt  = wk16 + qkv_elems;

    // 1024 mtiles * 24 tiles = 24576 waves; 8 waves/block -> 3072 blocks
    proj_kernel<<<3072, 256, 0, stream>>>(x, Wq, Wk, Wv, wq16, wk16, wvt);

    // 512 query tiles of 32 rows, one wave each
    attn_kernel<<<512, 32, 0, stream>>>(wq16, wk16, wvt, out);
}
